// GNN_Combined_1322849927570
// MI455X (gfx1250) — matrix operand order repaired and checked by reading, not compiled, not run on hardware
//
#include <hip/hip_runtime.h>
#include <hip/hip_bf16.h>

// ---------------- constants from the reference ----------------
#define NTOK   8000
#define INDIM  300
#define HIDW   128
#define HEADS1 4
#define OUTDIM 128
#define CLS    4
#define BSZ    32
#define NC     40
#define SEQ    60
#define NT     (BSZ * NC)   // 1280
#define HLSTM  100
#define GATE   (4 * HLSTM)  // 400

typedef __attribute__((ext_vector_type(16))) _Float16 v16h;
typedef __attribute__((ext_vector_type(8)))  _Float16 v8h;
typedef __attribute__((ext_vector_type(8)))  float    v8f;

// ---------------- small helpers ----------------
__device__ __forceinline__ float lrelu02(float x) { return x > 0.0f ? x : 0.2f * x; }
__device__ __forceinline__ float sigmoidf_(float x) { return 1.0f / (1.0f + __expf(-x)); }

// monotonic float<->uint encoding for atomicMax on floats
__device__ __forceinline__ unsigned encF(float f) {
    unsigned u = __float_as_uint(f);
    return (u & 0x80000000u) ? ~u : (u | 0x80000000u);
}
__device__ __forceinline__ float decF(unsigned u) {
    unsigned v = (u & 0x80000000u) ? (u & 0x7fffffffu) : ~u;
    return __uint_as_float(v);
}

// ---------------- fills ----------------
__global__ void fill_f32(float* p, float v, long n) {
    long i = (long)blockIdx.x * blockDim.x + threadIdx.x;
    if (i < n) p[i] = v;
}
__global__ void fill_u32(unsigned* p, unsigned v, long n) {
    long i = (long)blockIdx.x * blockDim.x + threadIdx.x;
    if (i < n) p[i] = v;
}

// =====================================================================
// WMMA GEMM: C[M,N] = A[M,K] @ B[K,N] (+bias)(+relu), N multiple of 128.
// Block tile 128(M) x 128(N), 256 threads = 8 wave32.
// Wave w owns rows [w*16, w*16+16) x all 128 cols: 8 WMMA accumulators
// sharing a single A fragment per 32-wide K step.
// A staged K-major (16B-aligned rows), B staged TRANSPOSED (N-major) so
// every fragment load is a contiguous ds_load_b128 pair.
// =====================================================================
#define BM   128
#define BN   128
#define BK   32
#define LDA  40   // halfs; 80B row stride -> 16B aligned
#define LDB  40

__global__ void __launch_bounds__(256)
wmma_gemm128(const float* __restrict__ A, const float* __restrict__ B,
             const float* __restrict__ bias, float* __restrict__ C,
             int M, int N, int K, int relu)
{
    __shared__ _Float16 As[BM * LDA];   // 10240 B
    __shared__ _Float16 BsT[BN * LDB];  // 10240 B

    const int bm   = blockIdx.y * BM;
    const int bn   = blockIdx.x * BN;
    const int tid  = threadIdx.x;
    const int w    = tid >> 5;
    const int lane = tid & 31;

    const int l15  = lane & 15;
    const int hi   = lane >> 4;          // 0 or 1
    const int arow = w * 16 + l15;       // A fragment row within block tile
    const int kh   = hi * 8;             // A K-half select
    const int kr16 = hi * 16;            // unused: B fragment also selects with kh

    v8f acc[8] = {};

    for (int k0 = 0; k0 < K; k0 += BK) {
        // prefetch next K-slab of A (streams token_graph once through HBM)
        if (tid < BM && k0 + BK < K && bm + tid < M)
            __builtin_prefetch(&A[(long)(bm + tid) * K + k0 + BK], 0, 1);

        const bool interior = (bm + BM <= M) && (k0 + BK <= K);
        if (interior) {
            // ---- A: 128x32 f32, 4 x float4 per thread, coalesced ----
#pragma unroll
            for (int j = 0; j < 4; ++j) {
                int idx = tid + j * 256;          // 0..1023
                int r   = idx >> 3;               // 0..127
                int c4  = (idx & 7) * 4;          // 0..28
                float4 v = *(const float4*)&A[(long)(bm + r) * K + k0 + c4];
                _Float16* d = &As[r * LDA + c4];
                d[0] = (_Float16)v.x; d[1] = (_Float16)v.y;
                d[2] = (_Float16)v.z; d[3] = (_Float16)v.w;
            }
            // ---- B: 32x128 f32, 4 x float4 per thread, store transposed ----
#pragma unroll
            for (int j = 0; j < 4; ++j) {
                int idx = tid + j * 256;          // 0..1023
                int kr  = idx >> 5;               // 0..31
                int c4  = (idx & 31) * 4;         // 0..124
                float4 v = *(const float4*)&B[(long)(k0 + kr) * N + bn + c4];
                BsT[(c4 + 0) * LDB + kr] = (_Float16)v.x;
                BsT[(c4 + 1) * LDB + kr] = (_Float16)v.y;
                BsT[(c4 + 2) * LDB + kr] = (_Float16)v.z;
                BsT[(c4 + 3) * LDB + kr] = (_Float16)v.w;
            }
        } else {
            // ---- guarded per-element path for M/K edges, zero-filled ----
            for (int i = tid; i < BM * BK; i += 256) {
                int r = i >> 5, c = i & 31;
                int gr = bm + r, gk = k0 + c;
                float av = (gr < M && gk < K) ? A[(long)gr * K + gk] : 0.0f;
                As[r * LDA + c] = (_Float16)av;
            }
            for (int i = tid; i < BK * BN; i += 256) {
                int kr = i >> 7, c = i & 127;
                int gk = k0 + kr;
                float bv = (gk < K) ? B[(long)gk * N + bn + c] : 0.0f;
                BsT[c * LDB + kr] = (_Float16)bv;
            }
        }
        __syncthreads();

        // ---- A fragment: two contiguous 16B runs (ds_load_b128 x2) ----
        const _Float16* ap = &As[arow * LDA];
        v8h a0 = *(const v8h*)(ap + kh);
        v8h a1 = *(const v8h*)(ap + 16 + kh);
        v16h a;
#pragma unroll
        for (int i = 0; i < 8; ++i) { a[i] = a0[i]; a[i + 8] = a1[i]; }

        // ---- 8 column tiles: two 16B runs each ----
#pragma unroll
        for (int ct = 0; ct < 8; ++ct) {
            const _Float16* bp = &BsT[(ct * 16 + l15) * LDB];
            v8h b0 = *(const v8h*)(bp + kh);
            v8h b1 = *(const v8h*)(bp + 16 + kh);
            v16h b;
#pragma unroll
            for (int i = 0; i < 8; ++i) { b[i] = b0[i]; b[i + 8] = b1[i]; }

            acc[ct] = __builtin_amdgcn_wmma_f32_16x16x32_f16(
                false, a, false, b, (short)0, acc[ct], false, false);
        }
        __syncthreads();
    }

    // C/D layout: VGPR r holds M = hi*8 + r, N = l15 within each 16x16 tile
    const int rbase = bm + w * 16 + hi * 8;
#pragma unroll
    for (int ct = 0; ct < 8; ++ct) {
        int col = bn + ct * 16 + l15;
        float bb = bias ? bias[col] : 0.0f;
#pragma unroll
        for (int r = 0; r < 8; ++r) {
            int row = rbase + r;
            if (row < M) {
                float v = acc[ct][r] + bb;
                if (relu) v = fmaxf(v, 0.0f);
                C[(long)row * N + col] = v;
            }
        }
    }
}

// ---------------- GAT pieces ----------------
__global__ void gat_el_er(const float* __restrict__ Z,
                          const float* __restrict__ al, const float* __restrict__ ar,
                          float* __restrict__ el, float* __restrict__ er,
                          int n, int heads, int hd)
{
    int idx = blockIdx.x * blockDim.x + threadIdx.x;
    if (idx >= n * heads) return;
    int node = idx / heads, h = idx % heads;
    const float* z = Z + (long)node * heads * hd + (long)h * hd;
    const float* a = al + (long)h * hd;
    const float* r = ar + (long)h * hd;
    float sl = 0.0f, sr = 0.0f;
    for (int d = 0; d < hd; ++d) { sl += z[d] * a[d]; sr += z[d] * r[d]; }
    el[idx] = sl; er[idx] = sr;
}

__global__ void gat_edge_max(const float* __restrict__ el, const float* __restrict__ er,
                             const int* __restrict__ src, const int* __restrict__ dst,
                             float* __restrict__ EV, unsigned* __restrict__ m,
                             int E, int heads)
{
    int idx = blockIdx.x * blockDim.x + threadIdx.x;
    if (idx >= E * heads) return;
    int e = idx / heads, h = idx % heads;
    float v = lrelu02(el[src[e] * heads + h] + er[dst[e] * heads + h]);
    EV[idx] = v;
    atomicMax(&m[dst[e] * heads + h], encF(v));
}

__global__ void gat_edge_exp(const int* __restrict__ dst,
                             float* __restrict__ EV, const unsigned* __restrict__ m,
                             float* __restrict__ s, int E, int heads)
{
    int idx = blockIdx.x * blockDim.x + threadIdx.x;
    if (idx >= E * heads) return;
    int e = idx / heads, h = idx % heads;
    float ex = __expf(EV[idx] - decF(m[dst[e] * heads + h]));
    EV[idx] = ex;
    atomicAdd(&s[dst[e] * heads + h], ex);
}

__global__ void gat_scatter(const float* __restrict__ Z, const float* __restrict__ EV,
                            const float* __restrict__ s,
                            const int* __restrict__ src, const int* __restrict__ dst,
                            float* __restrict__ out, int E, int heads, int hd)
{
    long idx = (long)blockIdx.x * blockDim.x + threadIdx.x;
    long tot = (long)E * heads * hd;
    if (idx >= tot) return;
    int d   = (int)(idx % hd);
    int rem = (int)(idx / hd);
    int h   = rem % heads;
    int e   = rem / heads;
    float a = EV[e * heads + h] / s[dst[e] * heads + h];
    atomicAdd(&out[(long)dst[e] * heads * hd + (long)h * hd + d],
              Z[(long)src[e] * heads * hd + (long)h * hd + d] * a);
}

__global__ void bias_act(float* __restrict__ x, const float* __restrict__ b,
                         long n, int cols, int relu)
{
    long i = (long)blockIdx.x * blockDim.x + threadIdx.x;
    if (i >= n) return;
    float v = x[i] + b[i % cols];
    if (relu) v = fmaxf(v, 0.0f);
    x[i] = v;
}

// ---------------- gather + concat ----------------
__global__ void gather_concat(const float* __restrict__ TOK, const float* __restrict__ INST,
                              const int* __restrict__ gid, const int* __restrict__ lid,
                              float* __restrict__ X)
{
    long idx = (long)blockIdx.x * blockDim.x + threadIdx.x;
    long tot = (long)BSZ * SEQ * (2 * OUTDIM);
    if (idx >= tot) return;
    int d  = (int)(idx % (2 * OUTDIM));
    int bs = (int)(idx / (2 * OUTDIM));
    int b  = bs / SEQ;
    float v;
    if (d < OUTDIM) v = TOK[(long)gid[bs] * OUTDIM + d];
    else            v = INST[((long)b * NC + lid[bs]) * OUTDIM + (d - OUTDIM)];
    X[idx] = v;
}

// ---------------- LSTM (one block per batch element, sequential in time) ----------------
__global__ void __launch_bounds__(128)
lstm_dir_kernel(const float* __restrict__ X,    // [B, SEQ, ind]
                const float* __restrict__ Wih,  // [400, ind]
                const float* __restrict__ Whh,  // [400, 100]
                const float* __restrict__ bias, // [400]
                float* __restrict__ Y,          // [B, SEQ, ystride] or null
                int y_off, int ystride,
                float* __restrict__ Hfinal,     // [B, 100] or null
                int ind, int reverse)
{
    __shared__ float h[HLSTM], c[HLSTM], xt[2 * OUTDIM], g[GATE];
    const int b = blockIdx.x, tid = threadIdx.x, bdim = blockDim.x;

    for (int i = tid; i < HLSTM; i += bdim) { h[i] = 0.0f; c[i] = 0.0f; }
    __syncthreads();

    for (int step = 0; step < SEQ; ++step) {
        int t = reverse ? (SEQ - 1 - step) : step;
        const float* x = X + ((long)b * SEQ + t) * ind;
        for (int i = tid; i < ind; i += bdim) xt[i] = x[i];
        __syncthreads();

        for (int j = tid; j < GATE; j += bdim) {
            float acc = bias[j];
            const float* wi = Wih + (long)j * ind;
            for (int k = 0; k < ind; ++k) acc += xt[k] * wi[k];
            const float* wh = Whh + (long)j * HLSTM;
            for (int k = 0; k < HLSTM; ++k) acc += h[k] * wh[k];
            g[j] = acc;
        }
        __syncthreads();

        for (int j = tid; j < HLSTM; j += bdim) {
            float ig = sigmoidf_(g[j]);
            float fg = sigmoidf_(g[HLSTM + j]);
            float gv = tanhf(g[2 * HLSTM + j]);
            float og = sigmoidf_(g[3 * HLSTM + j]);
            float cn = fg * c[j] + ig * gv;
            float hn = og * tanhf(cn);
            c[j] = cn; h[j] = hn;
            if (Y) Y[((long)b * SEQ + t) * ystride + y_off + j] = hn;
        }
        __syncthreads();
    }
    if (Hfinal) for (int j = tid; j < HLSTM; j += bdim) Hfinal[(long)b * HLSTM + j] = h[j];
}

// ---------------- final FC ----------------
__global__ void fc_out(const float* __restrict__ HF, const float* __restrict__ HB,
                       const float* __restrict__ W, const float* __restrict__ bvec,
                       float* __restrict__ out)
{
    int idx = blockIdx.x * blockDim.x + threadIdx.x;
    if (idx >= BSZ * CLS) return;
    int b = idx / CLS, cc = idx % CLS;
    float acc = bvec[cc];
    for (int k = 0; k < HLSTM; ++k) acc += HF[b * HLSTM + k] * W[k * CLS + cc];
    for (int k = 0; k < HLSTM; ++k) acc += HB[b * HLSTM + k] * W[(HLSTM + k) * CLS + cc];
    out[idx] = acc;
}

// ---------------- host side ----------------
static inline dim3 gemm_grid(int M, int N) { return dim3(N / BN, (M + BM - 1) / BM); }
static inline int blks(long n, int t) { return (int)((n + t - 1) / t); }

extern "C" void kernel_launch(void* const* d_in, const int* in_sizes, int n_in,
                              void* d_out, int out_size, void* d_ws, size_t ws_size,
                              hipStream_t stream) {
    const float* x0       = (const float*)d_in[0];   // instance_batch_embs [NT, IN]
    const float* G        = (const float*)d_in[1];   // token_graph [NTOK, NTOK]
    const float* temb     = (const float*)d_in[2];   // token_embs [NTOK, IN]
    const float* gcn_W1   = (const float*)d_in[3];
    const float* gcn_b1   = (const float*)d_in[4];
    const float* gcn_W2   = (const float*)d_in[5];
    const float* gcn_b2   = (const float*)d_in[6];
    const float* gat1_W   = (const float*)d_in[7];
    const float* gat1_al  = (const float*)d_in[8];
    const float* gat1_ar  = (const float*)d_in[9];
    const float* gat1_b   = (const float*)d_in[10];
    const float* gat2_W   = (const float*)d_in[11];
    const float* gat2_al  = (const float*)d_in[12];
    const float* gat2_ar  = (const float*)d_in[13];
    const float* gat2_b   = (const float*)d_in[14];
    const float* l0f_Wih  = (const float*)d_in[15];
    const float* l0f_Whh  = (const float*)d_in[16];
    const float* l0f_b    = (const float*)d_in[17];
    const float* l0b_Wih  = (const float*)d_in[18];
    const float* l0b_Whh  = (const float*)d_in[19];
    const float* l0b_b    = (const float*)d_in[20];
    const float* l1f_Wih  = (const float*)d_in[21];
    const float* l1f_Whh  = (const float*)d_in[22];
    const float* l1f_b    = (const float*)d_in[23];
    const float* l1b_Wih  = (const float*)d_in[24];
    const float* l1b_Whh  = (const float*)d_in[25];
    const float* l1b_b    = (const float*)d_in[26];
    const float* fc_W     = (const float*)d_in[27];
    const float* fc_b     = (const float*)d_in[28];
    const int*   src      = (const int*)d_in[29];
    const int*   dst      = (const int*)d_in[30];
    const int*   lid      = (const int*)d_in[31];
    const int*   gid      = (const int*)d_in[32];
    float* out = (float*)d_out;
    const int E = in_sizes[29];

    // ---- workspace layout (floats) ----
    float* ws = (float*)d_ws;
    long o = 0;
    float* Z1   = ws + o; o += (long)NT * HEADS1 * HIDW;
    float* OUT1 = ws + o; o += (long)NT * HEADS1 * HIDW;
    float* Z2   = ws + o; o += (long)NT * OUTDIM;
    float* OUT2 = ws + o; o += (long)NT * OUTDIM;
    float* TMP1 = ws + o; o += (long)NTOK * HIDW;
    float* TM   = ws + o; o += (long)NTOK * HIDW;
    float* TMP2 = ws + o; o += (long)NTOK * OUTDIM;
    float* TOK  = ws + o; o += (long)NTOK * OUTDIM;
    float* X    = ws + o; o += (long)BSZ * SEQ * 2 * OUTDIM;
    float* Y    = ws + o; o += (long)BSZ * SEQ * 2 * HLSTM;
    float* HF   = ws + o; o += (long)BSZ * HLSTM;
    float* HB   = ws + o; o += (long)BSZ * HLSTM;
    float* el1  = ws + o; o += (long)NT * HEADS1;
    float* er1  = ws + o; o += (long)NT * HEADS1;
    float* s1   = ws + o; o += (long)NT * HEADS1;
    unsigned* m1 = (unsigned*)(ws + o); o += (long)NT * HEADS1;
    float* el2  = ws + o; o += NT;
    float* er2  = ws + o; o += NT;
    float* s2   = ws + o; o += NT;
    unsigned* m2 = (unsigned*)(ws + o); o += NT;
    float* EV1  = ws + o; o += (long)E * HEADS1;
    float* EV2  = ws + o; o += (long)E;
    (void)ws_size; (void)n_in; (void)out_size;

    // smallest nonzero key in the monotonic encoding: below every real value's
    // encoding; every node has a self-loop so the max is always replaced.
    const unsigned NEGINF = 0x00000001u;

    // ========== GAT layer 1 ==========
    wmma_gemm128<<<gemm_grid(NT, HEADS1 * HIDW), 256, 0, stream>>>(
        x0, gat1_W, nullptr, Z1, NT, HEADS1 * HIDW, INDIM, 0);

    gat_el_er<<<blks((long)NT * HEADS1, 256), 256, 0, stream>>>(
        Z1, gat1_al, gat1_ar, el1, er1, NT, HEADS1, HIDW);

    fill_f32<<<blks((long)NT * HEADS1 * HIDW, 256), 256, 0, stream>>>(OUT1, 0.0f, (long)NT * HEADS1 * HIDW);
    fill_f32<<<blks((long)NT * HEADS1, 256), 256, 0, stream>>>(s1, 0.0f, (long)NT * HEADS1);
    fill_u32<<<blks((long)NT * HEADS1, 256), 256, 0, stream>>>(m1, NEGINF, (long)NT * HEADS1);

    gat_edge_max<<<blks((long)E * HEADS1, 256), 256, 0, stream>>>(el1, er1, src, dst, EV1, m1, E, HEADS1);
    gat_edge_exp<<<blks((long)E * HEADS1, 256), 256, 0, stream>>>(dst, EV1, m1, s1, E, HEADS1);
    gat_scatter<<<blks((long)E * HEADS1 * HIDW, 256), 256, 0, stream>>>(
        Z1, EV1, s1, src, dst, OUT1, E, HEADS1, HIDW);
    bias_act<<<blks((long)NT * HEADS1 * HIDW, 256), 256, 0, stream>>>(
        OUT1, gat1_b, (long)NT * HEADS1 * HIDW, HEADS1 * HIDW, 1);

    // ========== GAT layer 2 (heads=1) ==========
    wmma_gemm128<<<gemm_grid(NT, OUTDIM), 256, 0, stream>>>(
        OUT1, gat2_W, nullptr, Z2, NT, OUTDIM, HEADS1 * HIDW, 0);

    gat_el_er<<<blks((long)NT, 256), 256, 0, stream>>>(Z2, gat2_al, gat2_ar, el2, er2, NT, 1, OUTDIM);

    fill_f32<<<blks((long)NT * OUTDIM, 256), 256, 0, stream>>>(OUT2, 0.0f, (long)NT * OUTDIM);
    fill_f32<<<blks((long)NT, 256), 256, 0, stream>>>(s2, 0.0f, (long)NT);
    fill_u32<<<blks((long)NT, 256), 256, 0, stream>>>(m2, NEGINF, (long)NT);

    gat_edge_max<<<blks((long)E, 256), 256, 0, stream>>>(el2, er2, src, dst, EV2, m2, E, 1);
    gat_edge_exp<<<blks((long)E, 256), 256, 0, stream>>>(dst, EV2, m2, s2, E, 1);
    gat_scatter<<<blks((long)E * OUTDIM, 256), 256, 0, stream>>>(Z2, EV2, s2, src, dst, OUT2, E, 1, OUTDIM);
    bias_act<<<blks((long)NT * OUTDIM, 256), 256, 0, stream>>>(
        OUT2, gat2_b, (long)NT * OUTDIM, OUTDIM, 0);

    // ========== GCN on dense token graph (the HBM-bound part) ==========
    wmma_gemm128<<<gemm_grid(NTOK, HIDW), 256, 0, stream>>>(temb, gcn_W1, nullptr, TMP1, NTOK, HIDW, INDIM, 0);
    // G streams once (256MB); TMP1 (4MB) re-reads hit L2
    wmma_gemm128<<<gemm_grid(NTOK, HIDW), 256, 0, stream>>>(G, TMP1, gcn_b1, TM, NTOK, HIDW, NTOK, 1);
    wmma_gemm128<<<gemm_grid(NTOK, OUTDIM), 256, 0, stream>>>(TM, gcn_W2, nullptr, TMP2, NTOK, OUTDIM, HIDW, 0);
    wmma_gemm128<<<gemm_grid(NTOK, OUTDIM), 256, 0, stream>>>(G, TMP2, gcn_b2, TOK, NTOK, OUTDIM, NTOK, 0);

    // ========== gather + concat ==========
    gather_concat<<<blks((long)BSZ * SEQ * 2 * OUTDIM, 256), 256, 0, stream>>>(TOK, OUT2, gid, lid, X);

    // ========== BiLSTM stack ==========
    lstm_dir_kernel<<<BSZ, 128, 0, stream>>>(X, l0f_Wih, l0f_Whh, l0f_b, Y, 0,     2 * HLSTM, nullptr, 2 * OUTDIM, 0);
    lstm_dir_kernel<<<BSZ, 128, 0, stream>>>(X, l0b_Wih, l0b_Whh, l0b_b, Y, HLSTM, 2 * HLSTM, nullptr, 2 * OUTDIM, 1);
    lstm_dir_kernel<<<BSZ, 128, 0, stream>>>(Y, l1f_Wih, l1f_Whh, l1f_b, nullptr, 0, 0, HF, 2 * HLSTM, 0);
    lstm_dir_kernel<<<BSZ, 128, 0, stream>>>(Y, l1b_Wih, l1b_Whh, l1b_b, nullptr, 0, 0, HB, 2 * HLSTM, 1);

    // ========== final FC ==========
    fc_out<<<blks(BSZ * CLS, 128), 128, 0, stream>>>(HF, HB, fc_W, fc_b, out);
}
